// HyperGANUpBlock_47287589929740
// MI455X (gfx1250) — matrix operand order repaired and checked by reading, not compiled, not run on hardware
//
#include <hip/hip_runtime.h>
#include <hip/hip_bf16.h>

typedef __attribute__((ext_vector_type(16))) __bf16 v16bf;
typedef __attribute__((ext_vector_type(8)))  float  v8f;

#define B_    4
#define DIN   8
#define HIN   32
#define WIN   32
#define CIN   128
#define CMID  64
#define DOUT  16
#define HOUT  64
#define WOUT  64
#define K1    (27*CIN)               /* 3456 */
#define KSTEPS 108
#define WQ_ELEMS (KSTEPS*2048)       /* per-sample swizzled weight elems */
#define ROWS  (B_*DOUT*HOUT*WOUT)    /* 262144 rows of 64 channels */
#define NELEM (ROWS*CMID)            /* 16777216 */

__device__ __forceinline__ unsigned short f2bf(float f) {
  union { float f; unsigned u; } x; x.f = f;
  unsigned r = x.u + 0x7FFFu + ((x.u >> 16) & 1u);   // round-to-nearest-even
  return (unsigned short)(r >> 16);
}
__device__ __forceinline__ float bf2f(unsigned short h) {
  union { unsigned u; float f; } x; x.u = ((unsigned)h) << 16;
  return x.f;
}

// ---- CDNA5 async global->LDS copy (ASYNCcnt path), with safe fallback ----
typedef int vec4i __attribute__((vector_size(16)));
typedef __attribute__((address_space(1))) vec4i gvec4i;   // global
typedef __attribute__((address_space(3))) vec4i lvec4i;   // LDS

#if __has_builtin(__builtin_amdgcn_global_load_async_to_lds_b128)
#define ASYNC_LDS 1
__device__ __forceinline__ void cp16(const unsigned short* g, unsigned short* l) {
  __builtin_amdgcn_global_load_async_to_lds_b128(
      (gvec4i*)g, (lvec4i*)l, 0, 0);
}
__device__ __forceinline__ void cp_wait() {
  asm volatile("s_wait_asynccnt 0" ::: "memory");
}
#else
#define ASYNC_LDS 0
__device__ __forceinline__ void cp16(const unsigned short* g, unsigned short* l) {
  *(uint4*)l = *(const uint4*)g;
}
__device__ __forceinline__ void cp_wait() {}
#endif

// ---------------------------------------------------------------------------
// fp32 -> bf16 bulk convert (8 elems / thread)
// ---------------------------------------------------------------------------
__global__ __launch_bounds__(256) void cvt_bf16_8(
    const float* __restrict__ in, unsigned short* __restrict__ out, int n8)
{
  int i = blockIdx.x*blockDim.x + threadIdx.x;
  if (i >= n8) return;
  const float4* p = (const float4*)(in + (size_t)i*8);
  float4 a = p[0], b = p[1];
  uint4 u;
  u.x = (unsigned)f2bf(a.x) | ((unsigned)f2bf(a.y) << 16);
  u.y = (unsigned)f2bf(a.z) | ((unsigned)f2bf(a.w) << 16);
  u.z = (unsigned)f2bf(b.x) | ((unsigned)f2bf(b.y) << 16);
  u.w = (unsigned)f2bf(b.z) | ((unsigned)f2bf(b.w) << 16);
  *(uint4*)(out + (size_t)i*8) = u;
}

// ---------------------------------------------------------------------------
// Weights fp32 [b][K][64] -> bf16 in per-K-step WMMA fragment order:
//   wq[b][step][(g*32+lane)*16+slot] = w[b][step*32+k][g*16+nn],
//   lane = ((k>>3)&1)*16+nn, slot = (k>>4)*8 + (k&7).
// ---------------------------------------------------------------------------
__global__ __launch_bounds__(256) void prep_w(
    const float* __restrict__ w, unsigned short* __restrict__ wq, int total)
{
  int idx = blockIdx.x*blockDim.x + threadIdx.x;
  if (idx >= total) return;
  int b    = idx / WQ_ELEMS;
  int r    = idx % WQ_ELEMS;
  int step = r >> 11;
  int o    = r & 2047;
  int g    = o >> 9;
  int lane = (o >> 4) & 31;
  int slot = o & 15;
  int k    = ((slot >> 3) << 4) + ((lane >> 4) << 3) + (slot & 7);
  int n    = g*16 + (lane & 15);
  float v  = w[(size_t)b*(K1*CMID) + (size_t)(step*32 + k)*CMID + n];
  wq[idx]  = f2bf(v);
}

// ---------------------------------------------------------------------------
// Implicit-GEMM conv via WMMA. Block 128 thr (4 waves): tile M=64 (W row) x
// N=64, K-step=32 bf16. Activations: bf16, C=128 innermost. Weights: bf16,
// fragment-preswizzled. All staging = contiguous 16B chunks (async->LDS).
// UP: source grid is the pre-upsample grid (index >>1).
// ---------------------------------------------------------------------------
template <bool UP>
__global__ __launch_bounds__(128) void conv_wmma(
    const unsigned short* __restrict__ act,   // bf16 activations
    const unsigned short* __restrict__ wq,    // bf16 swizzled weights
    const unsigned short* __restrict__ zbuf,  // >=16 bf16 zeros
    unsigned short* __restrict__ ybf,         // bf16 out (conv1)
    float* __restrict__ yf)                   // fp32 out (conv2)
{
  __shared__ __align__(32) unsigned short sA[64*32];
  __shared__ __align__(32) unsigned short sB[32*64];

  const int tid    = threadIdx.x;
  const int lane   = tid & 31;
  const int waveId = tid >> 5;
  const int oh = blockIdx.x, od = blockIdx.y, b = blockIdx.z;

  const int m   = tid >> 1;            // tile row (= ow)
  const int k0  = (tid & 1) << 4;      // k half: 0 / 16
  const int mm  = m & 15, f = m >> 4;
  const int vb2 = (k0 >= 16) ? 8 : 0;  // vgpr-base*2 (halfword offset)
  // A chunks: k0..k0+7 -> lane-half h=0 ; k0+8..k0+15 -> h=1 (7.12.2 layout)
  const int ldsA0 = (f*32 +      mm)*16 + vb2;
  const int ldsA1 = (f*32 + 16 + mm)*16 + vb2;
  const int ldsB0 = tid * 16;

  const int SD = UP ? DIN : DOUT, SH = UP ? HIN : HOUT, SW = UP ? WIN : WOUT;
  const unsigned short* wqb = wq + (size_t)b * WQ_ELEMS;

  v8f acc[4];
#pragma unroll
  for (int j = 0; j < 4; ++j)
#pragma unroll
    for (int e = 0; e < 8; ++e) acc[j][e] = 0.0f;

  for (int step = 0; step < KSTEPS; ++step) {
    const int tap    = step >> 2;
    const int ciBase = (step & 3) << 5;
    const int kd = tap / 9, kh = (tap / 3) % 3, kw = tap % 3;

    const int idu = od + kd - 1, ihu = oh + kh - 1, iwu = m + kw - 1;
    const bool ok = ((unsigned)idu < 16u) && ((unsigned)ihu < 64u)
                 && ((unsigned)iwu < 64u);
    const int id = UP ? (idu >> 1) : idu;
    const int ih = UP ? (ihu >> 1) : ihu;
    const int iw = UP ? (iwu >> 1) : iwu;
    const unsigned short* src = ok
      ? act + (((size_t)(b*SD + id)*SH + ih)*SW + iw)*128 + ciBase + k0
      : zbuf;

    cp16(src,     &sA[ldsA0]);
    cp16(src + 8, &sA[ldsA1]);
    const unsigned short* wsrc = wqb + (size_t)step*2048 + ldsB0;
    cp16(wsrc,     &sB[ldsB0]);
    cp16(wsrc + 8, &sB[ldsB0 + 8]);

    cp_wait();
    __syncthreads();

    v16bf aFrag = *(const v16bf*)&sA[(waveId*32 + lane) << 4];
#pragma unroll
    for (int j = 0; j < 4; ++j) {
      v16bf bFrag = *(const v16bf*)&sB[(j*32 + lane) << 4];
      acc[j] = __builtin_amdgcn_wmma_f32_16x16x32_bf16(
          false, aFrag, false, bFrag, (short)0, acc[j], false, false);
    }

    __syncthreads();
  }

  // C layout: lane<16 -> M=v, lane>=16 -> M=v+8 ; N = lane%16.
  const size_t outBase = (((size_t)b*DOUT + od)*HOUT + oh)*WOUT*CMID;
  const int n = lane & 15, rh = lane >> 4;
  if (yf) {
#pragma unroll
    for (int j = 0; j < 4; ++j)
#pragma unroll
      for (int v = 0; v < 8; ++v) {
        int ow = waveId*16 + v + 8*rh;
        yf[outBase + ow*CMID + j*16 + n] = acc[j][v];
      }
  } else {
#pragma unroll
    for (int j = 0; j < 4; ++j)
#pragma unroll
      for (int v = 0; v < 8; ++v) {
        int ow = waveId*16 + v + 8*rh;
        ybf[outBase + ow*CMID + j*16 + n] = f2bf(acc[j][v]);
      }
  }
}

// ---------------------------------------------------------------------------
// BN statistics (training mode): per-channel sum / sumsq.
// ---------------------------------------------------------------------------
__global__ __launch_bounds__(256) void reduce_sums_bf(
    const unsigned short* __restrict__ y, float* __restrict__ sum,
    float* __restrict__ sq)
{
  const int tid = threadIdx.x, c = tid & 63, rg = tid >> 6;
  float s = 0.f, q = 0.f;
  for (int row = blockIdx.x*4 + rg; row < ROWS; row += gridDim.x*4) {
    float v = bf2f(y[(size_t)row*CMID + c]);
    s += v; q += v*v;
  }
  __shared__ float ss[256], qq[256];
  ss[tid] = s; qq[tid] = q;
  __syncthreads();
  if (tid < 64) {
    s = ss[tid] + ss[tid+64] + ss[tid+128] + ss[tid+192];
    q = qq[tid] + qq[tid+64] + qq[tid+128] + qq[tid+192];
    atomicAdd(&sum[c], s);
    atomicAdd(&sq[c],  q);
  }
}

__global__ __launch_bounds__(256) void reduce_sums_f32(
    const float* __restrict__ y, float* __restrict__ sum, float* __restrict__ sq)
{
  const int tid = threadIdx.x, c = tid & 63, rg = tid >> 6;
  float s = 0.f, q = 0.f;
  for (int row = blockIdx.x*4 + rg; row < ROWS; row += gridDim.x*4) {
    float v = y[(size_t)row*CMID + c];
    s += v; q += v*v;
  }
  __shared__ float ss[256], qq[256];
  ss[tid] = s; qq[tid] = q;
  __syncthreads();
  if (tid < 64) {
    s = ss[tid] + ss[tid+64] + ss[tid+128] + ss[tid+192];
    q = qq[tid] + qq[tid+64] + qq[tid+128] + qq[tid+192];
    atomicAdd(&sum[c], s);
    atomicAdd(&sq[c],  q);
  }
}

__global__ void finalize_bn(const float* __restrict__ sum, const float* __restrict__ sq,
                            const float* __restrict__ gamma, const float* __restrict__ beta,
                            float* __restrict__ scale, float* __restrict__ shift)
{
  const int c = threadIdx.x;
  const float cnt = (float)ROWS;
  float mean = sum[c] / cnt;
  float var  = sq[c] / cnt - mean*mean;
  float sc   = gamma[c] * rsqrtf(var + 1e-3f);
  scale[c] = sc;
  shift[c] = fmaf(-mean, sc, beta[c]);
}

// ---------------------------------------------------------------------------
// act1[pos][0:64]  = bf16(relu(bn1(y1)))   (from bf16 y1)
// act1[pos][64:128]= bf16(skip)            (from fp32 skip)
// 8 channels per thread.
// ---------------------------------------------------------------------------
__global__ __launch_bounds__(256) void build_act1(
    const unsigned short* __restrict__ y1bf, const float* __restrict__ skip,
    const float* __restrict__ sc, const float* __restrict__ sh,
    unsigned short* __restrict__ act, int n8)
{
  int i = blockIdx.x*blockDim.x + threadIdx.x;
  if (i >= n8) return;
  const size_t pos = (size_t)(i >> 4);
  const int c0 = (i & 15) * 8;
  float v[8];
  if (c0 < CMID) {
    uint4 u = *(const uint4*)(y1bf + pos*CMID + c0);
    unsigned uu[4] = {u.x, u.y, u.z, u.w};
#pragma unroll
    for (int p = 0; p < 4; ++p) {
      float lo = bf2f((unsigned short)(uu[p] & 0xFFFF));
      float hi = bf2f((unsigned short)(uu[p] >> 16));
      v[2*p+0] = fmaxf(0.f, fmaf(sc[c0+2*p+0], lo, sh[c0+2*p+0]));
      v[2*p+1] = fmaxf(0.f, fmaf(sc[c0+2*p+1], hi, sh[c0+2*p+1]));
    }
  } else {
    const float4* p4 = (const float4*)(skip + pos*CMID + (c0 - CMID));
    float4 a = p4[0], b = p4[1];
    v[0]=a.x; v[1]=a.y; v[2]=a.z; v[3]=a.w;
    v[4]=b.x; v[5]=b.y; v[6]=b.z; v[7]=b.w;
  }
  uint4 u;
  u.x = (unsigned)f2bf(v[0]) | ((unsigned)f2bf(v[1]) << 16);
  u.y = (unsigned)f2bf(v[2]) | ((unsigned)f2bf(v[3]) << 16);
  u.z = (unsigned)f2bf(v[4]) | ((unsigned)f2bf(v[5]) << 16);
  u.w = (unsigned)f2bf(v[6]) | ((unsigned)f2bf(v[7]) << 16);
  *(uint4*)(act + pos*128 + c0) = u;
}

__global__ __launch_bounds__(256) void bn_relu_inplace(
    float* __restrict__ y, const float* __restrict__ scale,
    const float* __restrict__ shift, int n4)
{
  int i = blockIdx.x*blockDim.x + threadIdx.x;
  if (i >= n4) return;
  float4 v = ((float4*)y)[i];
  int c = (i*4) & 63;
  v.x = fmaxf(0.f, fmaf(scale[c+0], v.x, shift[c+0]));
  v.y = fmaxf(0.f, fmaf(scale[c+1], v.y, shift[c+1]));
  v.z = fmaxf(0.f, fmaf(scale[c+2], v.z, shift[c+2]));
  v.w = fmaxf(0.f, fmaf(scale[c+3], v.w, shift[c+3]));
  ((float4*)y)[i] = v;
}

// ---------------------------------------------------------------------------
extern "C" void kernel_launch(void* const* d_in, const int* in_sizes, int n_in,
                              void* d_out, int out_size, void* d_ws, size_t ws_size,
                              hipStream_t stream) {
  (void)in_sizes; (void)n_in; (void)out_size; (void)ws_size;
  const float* x    = (const float*)d_in[0];
  const float* w1   = (const float*)d_in[1];
  const float* w2   = (const float*)d_in[2];
  const float* skip = (const float*)d_in[3];
  const float* g1   = (const float*)d_in[4];
  const float* be1  = (const float*)d_in[5];
  const float* g2   = (const float*)d_in[6];
  const float* be2  = (const float*)d_in[7];

  unsigned char* wsb = (unsigned char*)d_ws;
  float* stats = (float*)wsb;                    // 512 floats + zero line
  float* sum1 = stats;        float* sq1 = stats + 64;
  float* sum2 = stats + 128;  float* sq2 = stats + 192;
  float* sc1  = stats + 256;  float* sh1 = stats + 320;
  float* sc2  = stats + 384;  float* sh2 = stats + 448;
  unsigned short* zbuf = (unsigned short*)(stats + 512);   // 64B of bf16 zeros
  unsigned short* xbf  = (unsigned short*)(wsb + 4096);    // 4,194,304
  unsigned short* w1q  = xbf  + (size_t)4194304;           // 884,736
  unsigned short* w2q  = w1q  + (size_t)WQ_ELEMS*B_;       // 884,736
  unsigned short* y1bf = w2q  + (size_t)WQ_ELEMS*B_;       // 16,777,216
  unsigned short* act1 = y1bf + (size_t)NELEM;             // 33,554,432

  (void)hipMemsetAsync(wsb, 0, 4096, stream);    // zero sums + zero line

  // Pre-convert / pre-swizzle (once per call; tiny vs conv cost).
  cvt_bf16_8<<<(524288 + 255)/256, 256, 0, stream>>>(x, xbf, 524288);
  const int wtot = B_ * WQ_ELEMS;
  prep_w<<<(wtot + 255)/256, 256, 0, stream>>>(w1, w1q, wtot);
  prep_w<<<(wtot + 255)/256, 256, 0, stream>>>(w2, w2q, wtot);

  dim3 grid(HOUT, DOUT, B_);
  conv_wmma<true><<<grid, 128, 0, stream>>>(xbf, w1q, zbuf, y1bf, nullptr);
  reduce_sums_bf<<<1024, 256, 0, stream>>>(y1bf, sum1, sq1);
  finalize_bn<<<1, 64, 0, stream>>>(sum1, sq1, g1, be1, sc1, sh1);

  const int n8 = ROWS * 128 / 8;                 // 4,194,304
  build_act1<<<(n8 + 255)/256, 256, 0, stream>>>(y1bf, skip, sc1, sh1, act1, n8);

  conv_wmma<false><<<grid, 128, 0, stream>>>(act1, w2q, zbuf, nullptr, (float*)d_out);
  reduce_sums_f32<<<1024, 256, 0, stream>>>((const float*)d_out, sum2, sq2);
  finalize_bn<<<1, 64, 0, stream>>>(sum2, sq2, g2, be2, sc2, sh2);
  const int n4 = NELEM / 4;
  bn_relu_inplace<<<(n4 + 255)/256, 256, 0, stream>>>((float*)d_out, sc2, sh2, n4);
}
